// MoME_27736898797586
// MI455X (gfx1250) — matrix operand order repaired and checked by reading, not compiled, not run on hardware
//
#include <hip/hip_runtime.h>
#include <hip/hip_bf16.h>
#include <cstdint>

// ---------------------------------------------------------------------------
// MMoE forward on gfx1250.
// Grouped GEMMs (73 + 34 GFLOP) via v_wmma_f32_16x16x32_bf16 with:
//   * B tiles staged in LDS through global_load_async_to_lds_b128 (ASYNCcnt),
//     THREE-buffer pipeline with partial s_wait_asynccnt 2 so each copy has
//     two K-steps of WMMA work to hide under
//   * A fragments register double-buffered (b128 global loads)
//   * all 4 B fragments ds-loaded into distinct registers before the 8 WMMAs
//     so s_wait_dscnt is partial instead of a full drain per fragment
//   * 32x64 wave tile (8 WMMAs / K-step), 128x128 block tile, 8 waves
// Tail (gates/softmax/combine/towers) fused into one per-row kernel.
// ---------------------------------------------------------------------------

typedef __attribute__((ext_vector_type(16))) __bf16 v16bf;
typedef __attribute__((ext_vector_type(8)))  __bf16 v8bf;
typedef __attribute__((ext_vector_type(8)))  float  v8f;

#define B_ROWS 16384
#define NF     16
#define VOCAB  100000
#define ED     32
#define NNUM   13
#define NEXP   8
#define NTASK  2
#define H0_    512
#define H1_    256
#define TH_    128
#define D0_    525
#define K0PAD  544
#define EPSV   1e-5f

#define BLK_M  128
#define BLK_N  128
#define BK     32

// ---------------------------------------------------------------------------
// 1) Embedding gather + concat -> fp32 x [B,525] and zero-padded bf16 x [B,544]
// ---------------------------------------------------------------------------
__global__ __launch_bounds__(256)
void embed_concat(const int* __restrict__ cat, const float* __restrict__ num,
                  const float* __restrict__ emb,
                  float* __restrict__ xf, __bf16* __restrict__ xb) {
    const int b = blockIdx.x;
    for (int d = threadIdx.x; d < K0PAD; d += 256) {
        float v = 0.f;
        if (d < NF * ED) {
            const int f = d >> 5, c = d & 31;
            const long long row = (long long)cat[b * NF + f] + (long long)f * VOCAB;
            v = emb[row * ED + c];
        } else if (d < D0_) {
            v = num[b * NNUM + (d - NF * ED)];
        }
        if (d < D0_) xf[(size_t)b * D0_ + d] = v;
        xb[(size_t)b * K0PAD + d] = (__bf16)v;
    }
}

// ---------------------------------------------------------------------------
// 2) Weight prep: W [E][Korig][N] fp32 -> Wt [E][N][Kpad] bf16 (K zero-padded)
// ---------------------------------------------------------------------------
__global__ __launch_bounds__(256)
void prep_w(const float* __restrict__ W, __bf16* __restrict__ Wt,
            int Korig, int Kpad, int N, int nE) {
    const size_t total = (size_t)nE * N * Kpad;
    for (size_t i = (size_t)blockIdx.x * blockDim.x + threadIdx.x; i < total;
         i += (size_t)gridDim.x * blockDim.x) {
        const int    k = (int)(i % Kpad);
        const size_t r = i / Kpad;
        const int    n = (int)(r % N);
        const int    e = (int)(r / N);
        const float  v = (k < Korig) ? W[((size_t)e * Korig + k) * N + n] : 0.f;
        Wt[i] = (__bf16)v;
    }
}

// ---------------------------------------------------------------------------
// 3) Fold eval-mode BN + per-expert bias into scale[n] / shift[e][n]
// ---------------------------------------------------------------------------
__global__ void prep_bn(const float* __restrict__ g, const float* __restrict__ be,
                        const float* __restrict__ m, const float* __restrict__ v,
                        const float* __restrict__ bias, int N, int nE,
                        float* __restrict__ scale, float* __restrict__ shift) {
    const int n = blockIdx.x * blockDim.x + threadIdx.x;
    if (n >= N) return;
    const float s = g[n] * rsqrtf(v[n] + EPSV);
    scale[n] = s;
    const float base = be[n] - m[n] * s;
    for (int e = 0; e < nE; ++e)
        shift[e * N + n] = base + bias[e * N + n] * s;
}

// Async copy of 16B global -> LDS, tracked by ASYNCcnt (CDNA5).
__device__ __forceinline__ void async_b128(uint32_t lds_addr, const void* gaddr) {
    asm volatile("global_load_async_to_lds_b128 %0, %1, off"
                 :: "v"(lds_addr), "v"(gaddr) : "memory");
}
__device__ __forceinline__ void wait_async_le2() {
    asm volatile("s_wait_asynccnt 2" ::: "memory");
}

// ---------------------------------------------------------------------------
// 4) Grouped GEMM + fused BN/ReLU.
//    Block: 256 thr = 8 waves in 4(M) x 2(N); block tile 128x128.
//    Wave tile 32x64: 2 A frags x 4 B frags -> 8 wmma per K-step.
//    B tile (128 cols x 32 K bf16 = 8KB) triple-buffered in LDS via async;
//    copy of tile k+2 issued while computing tile k (2-deep pipeline).
// ---------------------------------------------------------------------------
template <typename OutT>
__global__ __launch_bounds__(256)
void wmma_gemm_bn_relu(const __bf16* __restrict__ A, size_t strideAe,
                       const __bf16* __restrict__ Bt,
                       int Kpad, int N, int M,
                       const float* __restrict__ scale,
                       const float* __restrict__ shift,
                       OutT* __restrict__ C) {
    __shared__ __bf16 Bs[3][BLK_N][BK];   // [buf][col][k] : 3 x 8KB

    const int e      = blockIdx.z;
    const int lane   = threadIdx.x & 31;
    const int wave   = threadIdx.x >> 5;
    const int half   = lane >> 4;
    const int l16    = lane & 15;
    const int wave_m = wave & 3;          // 4 strips of 32 rows
    const int wave_n = wave >> 2;         // 2 strips of 64 cols

    const int m0 = blockIdx.y * BLK_M;
    const int n0 = blockIdx.x * BLK_N;

    const __bf16* Ae  = A  + (size_t)e * strideAe;
    const __bf16* Bte = Bt + (size_t)e * (size_t)N * Kpad;

    // ---- async B-tile copy: thread t moves 32B (two b128, elements 0-7/8-15)
    const int cpCol  = threadIdx.x >> 1;          // 0..127
    const int cpPart = threadIdx.x & 1;           // 0 / 1
    const __bf16* cpSrc = Bte + (size_t)(n0 + cpCol) * Kpad + cpPart * 16;

    // rotating write targets (thread's slot) and read bases for 3 buffers
    uint32_t wA = (uint32_t)(uintptr_t)&Bs[0][cpCol][cpPart * 16];
    uint32_t wB = (uint32_t)(uintptr_t)&Bs[1][cpCol][cpPart * 16];
    uint32_t wC = (uint32_t)(uintptr_t)&Bs[2][cpCol][cpPart * 16];
    const __bf16* rA = &Bs[0][0][0];
    const __bf16* rB = &Bs[1][0][0];
    const __bf16* rC = &Bs[2][0][0];

    // per-lane column offset inside a B tile (elements)
    const int colOff = (wave_n * 64 + l16) * BK + half * 16;

    // ---- A fragment pointers: rows m0 + wave_m*32 + sub*16 + l16
    const __bf16* aRow0 = Ae + (size_t)(m0 + wave_m * 32 +  0 + l16) * Kpad + half * 16;
    const __bf16* aRow1 = Ae + (size_t)(m0 + wave_m * 32 + 16 + l16) * Kpad + half * 16;

    v8f acc[2][4];
#pragma unroll
    for (int s = 0; s < 2; ++s)
#pragma unroll
        for (int j = 0; j < 4; ++j) acc[s][j] = v8f{};

    // ---- prologue: tiles 0 and 1 in flight; A frags for k0=0
    async_b128(wA, cpSrc);
    async_b128(wA + 16, (const char*)cpSrc + 16);
    async_b128(wB, cpSrc + BK);
    async_b128(wB + 16, (const char*)(cpSrc + BK) + 16);
    v8bf aLo0 = *(const v8bf*)(aRow0);
    v8bf aHi0 = *(const v8bf*)(aRow0 + 8);
    v8bf aLo1 = *(const v8bf*)(aRow1);
    v8bf aHi1 = *(const v8bf*)(aRow1 + 8);
    wait_async_le2();            // tile 0 complete (in-order), tile 1 in flight
    __syncthreads();

    for (int k0 = 0; k0 < Kpad; k0 += BK) {
        // ---- issue async copy of tile k+2 into buffer C
        if (k0 + 2 * BK < Kpad) {
            const __bf16* s2 = cpSrc + (k0 + 2 * BK);
            async_b128(wC, s2);
            async_b128(wC + 16, (const char*)s2 + 16);
        }

        const bool more = (k0 + BK) < Kpad;

        // ---- issue next A fragment loads (register double buffer)
        v8bf nLo0, nHi0, nLo1, nHi1;
        if (more) {
            nLo0 = *(const v8bf*)(aRow0 + k0 + BK);
            nHi0 = *(const v8bf*)(aRow0 + k0 + BK + 8);
            nLo1 = *(const v8bf*)(aRow1 + k0 + BK);
            nHi1 = *(const v8bf*)(aRow1 + k0 + BK + 8);
        }

        v16bf a0 = __builtin_shufflevector(aLo0, aHi0,
                     0, 1, 2, 3, 4, 5, 6, 7, 8, 9, 10, 11, 12, 13, 14, 15);
        v16bf a1 = __builtin_shufflevector(aLo1, aHi1,
                     0, 1, 2, 3, 4, 5, 6, 7, 8, 9, 10, 11, 12, 13, 14, 15);

        // ---- all 4 B fragments from LDS first (8x ds_load_b128, distinct
        //      registers) so the scheduler can use partial s_wait_dscnt
        const __bf16* rd = rA + colOff;
        v16bf bf0 = *(const v16bf*)(rd);
        v16bf bf1 = *(const v16bf*)(rd + 16 * BK);
        v16bf bf2 = *(const v16bf*)(rd + 32 * BK);
        v16bf bf3 = *(const v16bf*)(rd + 48 * BK);

        acc[0][0] = __builtin_amdgcn_wmma_f32_16x16x32_bf16(
            false, a0, false, bf0, (short)0, acc[0][0], false, false);
        acc[1][0] = __builtin_amdgcn_wmma_f32_16x16x32_bf16(
            false, a1, false, bf0, (short)0, acc[1][0], false, false);
        acc[0][1] = __builtin_amdgcn_wmma_f32_16x16x32_bf16(
            false, a0, false, bf1, (short)0, acc[0][1], false, false);
        acc[1][1] = __builtin_amdgcn_wmma_f32_16x16x32_bf16(
            false, a1, false, bf1, (short)0, acc[1][1], false, false);
        acc[0][2] = __builtin_amdgcn_wmma_f32_16x16x32_bf16(
            false, a0, false, bf2, (short)0, acc[0][2], false, false);
        acc[1][2] = __builtin_amdgcn_wmma_f32_16x16x32_bf16(
            false, a1, false, bf2, (short)0, acc[1][2], false, false);
        acc[0][3] = __builtin_amdgcn_wmma_f32_16x16x32_bf16(
            false, a0, false, bf3, (short)0, acc[0][3], false, false);
        acc[1][3] = __builtin_amdgcn_wmma_f32_16x16x32_bf16(
            false, a1, false, bf3, (short)0, acc[1][3], false, false);

        aLo0 = nLo0; aHi0 = nHi0; aLo1 = nLo1; aHi1 = nHi1;

        if (more) {
            wait_async_le2();    // tile k+1 landed; tile k+2 still in flight
            __syncthreads();     // everyone done reading buffer A
            // rotate buffers: A <- B <- C <- A
            const uint32_t tw = wA; wA = wB; wB = wC; wC = tw;
            const __bf16* tr = rA; rA = rB; rB = rC; rC = tr;
        }
    }

    // ---- epilogue: D layout -> VGPR r is row (half*8 + r), col l16
#pragma unroll
    for (int s = 0; s < 2; ++s) {
#pragma unroll
        for (int j = 0; j < 4; ++j) {
            const int   n  = n0 + wave_n * 64 + j * 16 + l16;
            const float sc = scale[n];
            const float sh = shift[e * N + n];
#pragma unroll
            for (int r = 0; r < 8; ++r) {
                const int m = m0 + wave_m * 32 + s * 16 + half * 8 + r;
                float v = fmaf(acc[s][j][r], sc, sh);
                v = v > 0.f ? v : 0.f;
                C[((size_t)e * M + m) * N + n] = (OutT)v;
            }
        }
    }
}

// ---------------------------------------------------------------------------
// 5) Fused tail per row: gates -> softmax(E=8) -> combine -> tower -> sigmoid
// ---------------------------------------------------------------------------
__global__ __launch_bounds__(128)
void gate_tower(const float* __restrict__ xf, const float* __restrict__ eo,
                const float* __restrict__ Wg,  const float* __restrict__ bg,
                const float* __restrict__ Wt1, const float* __restrict__ bt1,
                const float* __restrict__ tg,  const float* __restrict__ tb,
                const float* __restrict__ tm,  const float* __restrict__ tv,
                const float* __restrict__ Wt2, const float* __restrict__ bt2,
                float* __restrict__ out) {
    __shared__ float xs[D0_];
    __shared__ float logits[NTASK * NEXP];
    __shared__ float gates[NTASK * NEXP];
    __shared__ float fea[NTASK * H1_];
    __shared__ float red[NTASK * TH_];

    const int b = blockIdx.x;
    const int h = threadIdx.x;

    for (int d = h; d < D0_; d += 128) xs[d] = xf[(size_t)b * D0_ + d];
    __syncthreads();

    if (h < NTASK * NEXP) {
        const int t = h >> 3, ex = h & 7;
        float acc = bg[t * NEXP + ex];
        for (int k = 0; k < D0_; ++k)
            acc = fmaf(xs[k], Wg[((size_t)t * D0_ + k) * NEXP + ex], acc);
        logits[h] = acc;
    }
    __syncthreads();

    if (h < NTASK) {
        float mx = -1e30f;
        for (int ex = 0; ex < NEXP; ++ex) mx = fmaxf(mx, logits[h * NEXP + ex]);
        float tmp[NEXP], sum = 0.f;
        for (int ex = 0; ex < NEXP; ++ex) {
            tmp[ex] = __expf(logits[h * NEXP + ex] - mx);
            sum += tmp[ex];
        }
        const float inv = 1.f / sum;
        for (int ex = 0; ex < NEXP; ++ex) gates[h * NEXP + ex] = tmp[ex] * inv;
    }
    __syncthreads();

#pragma unroll
    for (int t = 0; t < NTASK; ++t) {
        for (int k = h; k < H1_; k += 128) {
            float f = 0.f;
#pragma unroll
            for (int ex = 0; ex < NEXP; ++ex)
                f = fmaf(gates[t * NEXP + ex],
                         eo[((size_t)ex * B_ROWS + b) * H1_ + k], f);
            fea[t * H1_ + k] = f;
        }
    }
    __syncthreads();

#pragma unroll
    for (int t = 0; t < NTASK; ++t) {
        float acc = bt1[t * TH_ + h];
        for (int k = 0; k < H1_; ++k)
            acc = fmaf(fea[t * H1_ + k], Wt1[((size_t)t * H1_ + k) * TH_ + h], acc);
        float z = (acc - tm[t * TH_ + h]) * rsqrtf(tv[t * TH_ + h] + EPSV);
        z = z * tg[t * TH_ + h] + tb[t * TH_ + h];
        z = z > 0.f ? z : 0.f;
        red[t * TH_ + h] = z * Wt2[t * TH_ + h];
    }
    __syncthreads();

    for (int s = 64; s > 0; s >>= 1) {
        if (h < s) {
            red[h]       += red[h + s];
            red[TH_ + h] += red[TH_ + h + s];
        }
        __syncthreads();
    }
    if (h < NTASK) {
        const float o = red[h * TH_] + bt2[h];
        out[(size_t)h * B_ROWS + b] = 1.f / (1.f + __expf(-o));
    }
}

extern "C" void kernel_launch(void* const* d_in, const int* in_sizes, int n_in,
                              void* d_out, int out_size, void* d_ws, size_t ws_size,
                              hipStream_t stream) {
    const int*   cat  = (const int*)d_in[0];
    const float* num  = (const float*)d_in[1];
    // d_in[2] = epoch (harness fixes it >= M -> MoE branch); unused.
    const float* emb  = (const float*)d_in[3];
    const float* W0   = (const float*)d_in[4];
    const float* b0   = (const float*)d_in[5];
    const float* g0   = (const float*)d_in[6];
    const float* be0  = (const float*)d_in[7];
    const float* m0v  = (const float*)d_in[8];
    const float* v0   = (const float*)d_in[9];
    const float* W1   = (const float*)d_in[10];
    const float* b1   = (const float*)d_in[11];
    const float* g1   = (const float*)d_in[12];
    const float* be1  = (const float*)d_in[13];
    const float* m1v  = (const float*)d_in[14];
    const float* v1   = (const float*)d_in[15];
    const float* Wg   = (const float*)d_in[16];
    const float* bg   = (const float*)d_in[17];
    const float* Wt1  = (const float*)d_in[18];
    const float* bt1  = (const float*)d_in[19];
    const float* tg   = (const float*)d_in[20];
    const float* tb   = (const float*)d_in[21];
    const float* tm   = (const float*)d_in[22];
    const float* tv   = (const float*)d_in[23];
    const float* Wt2  = (const float*)d_in[24];
    const float* bt2  = (const float*)d_in[25];
    float* out = (float*)d_out;

    char* p = (char*)d_ws;
    auto take = [&](size_t bytes) -> void* {
        void* r = (void*)p;
        p += (bytes + 255) & ~(size_t)255;
        return r;
    };

    __bf16* xb     = (__bf16*)take((size_t)B_ROWS * K0PAD * sizeof(__bf16));
    float*  xf     = (float*)take((size_t)B_ROWS * D0_ * sizeof(float));
    __bf16* W0t    = (__bf16*)take((size_t)NEXP * H0_ * K0PAD * sizeof(__bf16));
    __bf16* W1t    = (__bf16*)take((size_t)NEXP * H1_ * H0_ * sizeof(__bf16));
    float*  scale0 = (float*)take((size_t)H0_ * sizeof(float));
    float*  shift0 = (float*)take((size_t)NEXP * H0_ * sizeof(float));
    float*  scale1 = (float*)take((size_t)H1_ * sizeof(float));
    float*  shift1 = (float*)take((size_t)NEXP * H1_ * sizeof(float));
    __bf16* h0buf  = (__bf16*)take((size_t)NEXP * B_ROWS * H0_ * sizeof(__bf16));
    float*  eobuf  = (float*)take((size_t)NEXP * B_ROWS * H1_ * sizeof(float));

    embed_concat<<<B_ROWS, 256, 0, stream>>>(cat, num, emb, xf, xb);
    prep_w<<<1024, 256, 0, stream>>>(W0, W0t, D0_, K0PAD, H0_, NEXP);
    prep_w<<<1024, 256, 0, stream>>>(W1, W1t, H0_, H0_, H1_, NEXP);
    prep_bn<<<(H0_ + 255) / 256, 256, 0, stream>>>(g0, be0, m0v, v0, b0, H0_, NEXP,
                                                   scale0, shift0);
    prep_bn<<<(H1_ + 255) / 256, 256, 0, stream>>>(g1, be1, m1v, v1, b1, H1_, NEXP,
                                                   scale1, shift1);

    // layer 0: x[B,544] x W0t[E,512,544] -> h0 bf16 [E,B,512]
    wmma_gemm_bn_relu<__bf16><<<dim3(H0_ / BLK_N, B_ROWS / BLK_M, NEXP), 256, 0,
                                stream>>>(
        xb, /*strideAe=*/0, W0t, K0PAD, H0_, B_ROWS, scale0, shift0, h0buf);

    // layer 1: h0[E,B,512] x W1t[E,256,512] -> eo fp32 [E,B,256]
    wmma_gemm_bn_relu<float><<<dim3(H1_ / BLK_N, B_ROWS / BLK_M, NEXP), 256, 0,
                               stream>>>(
        h0buf, /*strideAe=*/(size_t)B_ROWS * H0_, W1t, H0_, H1_, B_ROWS,
        scale1, shift1, eobuf);

    gate_tower<<<B_ROWS, 128, 0, stream>>>(xf, eobuf, Wg, bg, Wt1, bt1,
                                           tg, tb, tm, tv, Wt2, bt2, out);
}
